// SCGCN_self_72310069395817
// MI455X (gfx1250) — matrix operand order repaired and checked by reading, not compiled, not run on hardware
//
#include <hip/hip_runtime.h>
#include <hip/hip_bf16.h>
#include <math.h>

// ---------------------------------------------------------------------------
// SCGCN fused kernel for MI455X (gfx1250, wave32, WMMA)
//   B=256, C=1024, N=80
//   Phase 1: G = x^T x  (80x80, K=1024)  via v_wmma_f32_16x16x32_bf16
//   adj[n,m] = G[n,m] / (max(sqrt(G[n,n]),eps) * max(sqrt(G[m,m]),eps))
//   Phase 2: hT = adjT (80x80, K padded to 96) x  x_bf16 (K=80, N'=1024)
//            s[n] = sum_c lrelu(hT[n,c]) * w[c]  (in-register + LDS atomics)
//   Softmax(s/0.025)+1 in LDS, then out = x * attn (float4 streaming).
// One workgroup (256 threads = 8 waves) per batch element.
//
// Roofline: ~6.7 GFLOP of GEMM vs >=168 MB mandatory HBM traffic -> memory
// bound at 23.3 TB/s only if the matmuls run on the bf16 WMMA pipe (f32
// accumulate).  x (84 MB) fits in the 192 MB L2, so the three passes over x
// mostly hit L2.  global_prefetch_b8 overlaps next-chunk fetch with WMMA.
// ---------------------------------------------------------------------------

#define BATCH 256
#define CH    1024
#define NN    80
#define NEG_SLOPE 0.2f
#define INV_TEMP  40.0f          // 1 / 0.025
#define EPSN      1e-12f
#define KCH       256            // channel chunk staged in LDS per phase step

// LDS strides in bf16 elements: multiples of 8 (=> 16B-aligned b128 loads),
// chosen so row-to-row bank offsets avoid conflicts on fragment loads.
#define SXT_S 264                // xT chunk: 80 rows x 264   (phase 1)
#define XC_S  104                // x  chunk: 256 rows x 104  (phase 2, K pad 80->96 zeros)
#define ADJ_S 104                // adjT:     80 rows x 104   (K zero-padded)

// Dynamic-LDS layout (bytes); phase-2 XC buffer reuses the phase-1 region.
#define OFF_SXT   0
#define OFF_G     (80 * SXT_S * 2)            // 42240 (16B aligned)
#define OFF_BUFE  (OFF_G + 80 * 80 * 4)       // 67840
#define OFF_ADJ   OFF_BUFE                    // 67840
#define OFF_W     (OFF_ADJ + 80 * ADJ_S * 2)  // 84480
#define OFF_SACC  (OFF_W + CH * 4)            // 88576
#define OFF_RN    (OFF_SACC + 80 * 4)         // 88896
#define OFF_ATTN  (OFF_RN + 80 * 4)           // 89216
#define OFF_SCAL  (OFF_ATTN + 80 * 4)         // 89536
#define SMEM_BYTES (OFF_SCAL + 16)            // 89552  (~90KB < 320KB/WGP)

typedef __bf16 v16bf __attribute__((ext_vector_type(16)));
typedef float  v8f   __attribute__((ext_vector_type(8)));

__device__ __forceinline__ unsigned short f2bf(float f) {
    // round-to-nearest-even float -> bf16 (raw u16)
    unsigned u = __float_as_uint(f);
    unsigned r = u + 0x7FFFu + ((u >> 16) & 1u);
    return (unsigned short)(r >> 16);
}

// A-matrix fragment (16x32 bf16, M x K), ISA layout:
//  lanes 0-15: M=lane,  elems 0..7 -> K=K0+0..7,  elems 8..15 -> K=K0+16..23
//  lanes16-31: M=lane-16, elems     -> K=K0+8..15 and K0+24..31
__device__ __forceinline__ v16bf ldA(const unsigned short* p, int S, int R,
                                     int K0, int lr, int lh) {
    const unsigned short* q = p + (R + lr) * S + K0 + (lh ? 8 : 0);
    union { uint4 u[2]; v16bf v; } f;
    f.u[0] = *(const uint4*)(q);        // 8 bf16, 16B aligned
    f.u[1] = *(const uint4*)(q + 16);   // 8 bf16
    return f.v;
}

// B-matrix fragment (32x16 bf16, K x N), ISA layout:
//  lanes 0-15: N=lane, elems 0..7 -> K=K0+0..7, elems 8..15 -> K=K0+16..23
//  lanes16-31: K=K0+8..15 and K0+24..31 (same K order as ldA)
__device__ __forceinline__ v16bf ldB(const unsigned short* p, int S, int CB,
                                     int K0, int lr, int lh) {
    const unsigned short* q = p + (CB + lr) * S + K0 + (lh ? 8 : 0);
    union { uint4 u[2]; v16bf v; } f;
    f.u[0] = *(const uint4*)(q);
    f.u[1] = *(const uint4*)(q + 16);
    return f.v;
}

// Prefetch `bytes` of global memory spread over the workgroup (one byte per
// 128B cacheline per ISA prefetch semantics) -> global_prefetch_b8.
__device__ __forceinline__ void wg_prefetch(const char* base, int bytes, int tid) {
    int lines = bytes >> 7;
    for (int i = tid; i < lines; i += 256)
        __builtin_prefetch(base + (i << 7), 0, 1);
}

__global__ __launch_bounds__(256)
void scgcn_fused(const float* __restrict__ X, const float* __restrict__ W,
                 const float* __restrict__ Bb, float* __restrict__ Out) {
    extern __shared__ __align__(16) char smem[];
    unsigned short* sXT  = (unsigned short*)(smem + OFF_SXT);
    float*          sG   = (float*)(smem + OFF_G);
    unsigned short* sXC  = (unsigned short*)(smem + OFF_SXT);  // reuses phase-1 region
    unsigned short* sAdj = (unsigned short*)(smem + OFF_ADJ);
    float*          sW   = (float*)(smem + OFF_W);
    float*          sAcc = (float*)(smem + OFF_SACC);
    float*          sRn  = (float*)(smem + OFF_RN);
    float*          sAt  = (float*)(smem + OFF_ATTN);
    float*          sScal= (float*)(smem + OFF_SCAL);

    const int tid  = threadIdx.x;
    const int lane = tid & 31;
    const int wave = tid >> 5;
    const int lr   = lane & 15;
    const int lh   = (lane >> 4) & 1;

    const int b = blockIdx.x;
    const size_t xbase = (size_t)b * CH * NN;
    const float4* xv4 = (const float4*)(X + xbase);

    // init: s accumulator and the 1x1-conv weights into LDS
    if (tid < 80) sAcc[tid] = 0.0f;
    for (int i = tid; i < CH; i += 256) sW[i] = W[i];

    // warm the cache for chunk 0 while we still owe nothing to LDS
    wg_prefetch((const char*)(X + xbase), KCH * NN * 4, tid);

    // ---------------- Phase 1: G = x^T x (K-chunked over channels) ----------
    v8f accG[4];
#pragma unroll
    for (int s = 0; s < 4; ++s)
#pragma unroll
        for (int r = 0; r < 8; ++r) accG[s][r] = 0.0f;

    for (int kc = 0; kc < CH / KCH; ++kc) {
        __syncthreads();
        // stage xT chunk: sXT[n][c_local] <- bf16(x[b, kc*KCH + c, n])
        const float4* src = xv4 + (size_t)kc * KCH * (NN / 4);
        for (int i = tid; i < KCH * (NN / 4); i += 256) {
            int c  = i / (NN / 4);
            int n4 = (i % (NN / 4)) * 4;
            float4 v = src[i];
            sXT[(n4 + 0) * SXT_S + c] = f2bf(v.x);
            sXT[(n4 + 1) * SXT_S + c] = f2bf(v.y);
            sXT[(n4 + 2) * SXT_S + c] = f2bf(v.z);
            sXT[(n4 + 3) * SXT_S + c] = f2bf(v.w);
        }
        __syncthreads();
        // overlap: prefetch the next chunk while the WMMA pipe is busy
        if (kc + 1 < CH / KCH)
            wg_prefetch((const char*)(X + xbase + (size_t)(kc + 1) * KCH * NN),
                        KCH * NN * 4, tid);
        // 25 output tiles (5x5), round-robin over 8 waves; A row = n, B col = m,
        // both operands come from sXT rows (G[n,m] = sum_c xT[n,c] * xT[m,c]).
#pragma unroll
        for (int s = 0; s < 4; ++s) {
            int t = wave + 8 * s;               // wave-uniform guard: EXEC all-1s
            if (t < 25) {
                int ti = t / 5, tj = t % 5;
#pragma unroll
                for (int kk = 0; kk < KCH / 32; ++kk) {
                    v16bf a = ldA(sXT, SXT_S, 16 * ti, kk * 32, lr, lh);
                    v16bf bm = ldB(sXT, SXT_S, 16 * tj, kk * 32, lr, lh);
                    accG[s] = __builtin_amdgcn_wmma_f32_16x16x32_bf16(
                        false, a, false, bm, (short)0, accG[s], false, false);
                }
            }
        }
    }
    // store G tiles to LDS (D layout: VGPR r -> row r (+8 for lanes 16-31))
#pragma unroll
    for (int s = 0; s < 4; ++s) {
        int t = wave + 8 * s;
        if (t < 25) {
            int ti = t / 5, tj = t % 5;
#pragma unroll
            for (int r = 0; r < 8; ++r)
                sG[(16 * ti + r + lh * 8) * 80 + 16 * tj + lr] = accG[s][r];
        }
    }
    __syncthreads();

    // reciprocal norms from the diagonal, then adjT (symmetric) in bf16,
    // K zero-padded to 96 so phase 2 runs 3 clean K=32 steps.
    if (tid < 80) {
        float g = sG[tid * 80 + tid];
        sRn[tid] = 1.0f / fmaxf(sqrtf(g), EPSN);
    }
    __syncthreads();
    for (int idx = tid; idx < 80 * ADJ_S; idx += 256) {
        int m = idx / ADJ_S, k = idx % ADJ_S;
        float v = (k < 80) ? sG[m * 80 + k] * sRn[m] * sRn[k] : 0.0f;
        sAdj[idx] = f2bf(v);
    }

    // ---------------- Phase 2: hT = adjT x X, fold into s ------------------
    for (int cc = 0; cc < CH / KCH; ++cc) {
        __syncthreads();
        // stage x chunk as B operand: sXC[c_local][n], n padded 80->104, zeros past 80
        const float4* src = xv4 + (size_t)cc * KCH * (NN / 4);
        for (int i = tid; i < KCH * (NN / 4); i += 256) {
            int cl = i / (NN / 4);
            int n4 = (i % (NN / 4)) * 4;
            float4 v = src[i];
            unsigned a0 = (unsigned)f2bf(v.x) | ((unsigned)f2bf(v.y) << 16);
            unsigned a1 = (unsigned)f2bf(v.z) | ((unsigned)f2bf(v.w) << 16);
            *(uint2*)(&sXC[cl * XC_S + n4]) = make_uint2(a0, a1);
        }
        for (int i = tid; i < KCH * 12; i += 256) {  // zero pad n = 80..103
            int cl = i / 12, j = i % 12;
            *(unsigned*)(&sXC[cl * XC_S + 80 + 2 * j]) = 0u;
        }
        __syncthreads();
        if (cc + 1 < CH / KCH)
            wg_prefetch((const char*)(X + xbase + (size_t)(cc + 1) * KCH * NN),
                        KCH * NN * 4, tid);
        // 5 row-tiles x 16 col-tiles = 80 tiles; 10 per wave
        for (int s = 0; s < 10; ++s) {
            int t  = wave + 8 * s;      // bijective over 0..79
            int ti = t / 16, ct = t % 16;
            v8f acc;
#pragma unroll
            for (int r = 0; r < 8; ++r) acc[r] = 0.0f;
#pragma unroll
            for (int kt = 0; kt < 3; ++kt) {
                v16bf a  = ldA(sAdj, ADJ_S, 16 * ti, kt * 32, lr, lh);
                v16bf bx = ldB(sXC,  XC_S,  16 * ct, kt * 32, lr, lh);
                acc = __builtin_amdgcn_wmma_f32_16x16x32_bf16(
                    false, a, false, bx, (short)0, acc, false, false);
            }
            // lrelu(hT[m,c]) * w[c]; reduce the 16 lanes sharing row m, one atomic
            float wc = sW[cc * KCH + ct * 16 + lr];
#pragma unroll
            for (int r = 0; r < 8; ++r) {
                float v = acc[r];
                v = (v >= 0.0f) ? v : NEG_SLOPE * v;
                float contrib = v * wc;
                contrib += __shfl_xor(contrib, 1, 16);
                contrib += __shfl_xor(contrib, 2, 16);
                contrib += __shfl_xor(contrib, 4, 16);
                contrib += __shfl_xor(contrib, 8, 16);
                if (lr == 0)
                    atomicAdd(&sAcc[16 * ti + r + lh * 8], contrib);
            }
        }
    }
    __syncthreads();

    // final pass re-reads all of x; start pulling it toward the WGP now
    wg_prefetch((const char*)(X + xbase), CH * NN * 4, tid);

    // ---------------- softmax(s/TEMP) + 1 (N=80, trivial) -------------------
    const float bias = Bb[0];
    if (tid < 80) {
        float sv = sAcc[tid] + bias;
        sv = (sv >= 0.0f) ? sv : NEG_SLOPE * sv;
        sAt[tid] = sv;
    }
    __syncthreads();
    if (tid == 0) {
        float m = -3.4e38f;
        for (int n = 0; n < 80; ++n) m = fmaxf(m, sAt[n]);
        sScal[0] = m;
    }
    __syncthreads();
    if (tid < 80) sAt[tid] = __expf((sAt[tid] - sScal[0]) * INV_TEMP);
    __syncthreads();
    if (tid == 0) {
        float sum = 0.0f;
        for (int n = 0; n < 80; ++n) sum += sAt[n];
        sScal[1] = 1.0f / sum;
    }
    __syncthreads();
    if (tid < 80) sAt[tid] = sAt[tid] * sScal[1] + 1.0f;
    __syncthreads();

    // ---------------- out = x * attn (streaming, float4) --------------------
    float4* ov = (float4*)(Out + xbase);
    for (int i = tid; i < CH * (NN / 4); i += 256) {
        int n4 = (i % (NN / 4)) * 4;
        float4 v = xv4[i];
        v.x *= sAt[n4 + 0];
        v.y *= sAt[n4 + 1];
        v.z *= sAt[n4 + 2];
        v.w *= sAt[n4 + 3];
        ov[i] = v;
    }
}

extern "C" void kernel_launch(void* const* d_in, const int* in_sizes, int n_in,
                              void* d_out, int out_size, void* d_ws, size_t ws_size,
                              hipStream_t stream) {
    (void)in_sizes; (void)n_in; (void)out_size; (void)d_ws; (void)ws_size;
    const float* x = (const float*)d_in[0];
    const float* w = (const float*)d_in[1];
    const float* b = (const float*)d_in[2];
    float* out = (float*)d_out;

    // allow >64KB dynamic LDS (gfx1250 WGP supports 320KB); not a stream op,
    // deterministic every call.
    (void)hipFuncSetAttribute((const void*)scgcn_fused,
                              hipFuncAttributeMaxDynamicSharedMemorySize,
                              SMEM_BYTES);

    scgcn_fused<<<dim3(BATCH), dim3(256), SMEM_BYTES, stream>>>(x, w, b, out);
}
